// new_knn_81466939670828
// MI455X (gfx1250) — compile-verified, matrix operand order checked by reading, not run on hardware
//
#include <hip/hip_runtime.h>
#include <hip/hip_bf16.h>

// ---------------------------------------------------------------------------
// B=8, C=64, N=1000, K_TMP=50, K=20 ; M = K_TMP*C = 3200 tokens, feature 1000
// NP = 1024 zero-padded K dim -> WMMA K-loop (BK=32) has no tail.
// Heavy GEMMs (~560 GFLOP) run as v_wmma_f32_16x16x32_f16 with f32 accum.
// Global->LDS staging uses CDNA5 async copies (ASYNCcnt) with double buffering
// so tile t+1 streams into LDS while WMMA consumes tile t.
// ---------------------------------------------------------------------------

#define BB   8
#define CC   64
#define NN   1000
#define KT   50
#define KSEL 20
#define MM   3200      // KT*CC
#define NP   1024      // padded feature dim

typedef __attribute__((ext_vector_type(16))) _Float16 v16h;
typedef __attribute__((ext_vector_type(8)))  _Float16 f16x8;
typedef __attribute__((ext_vector_type(8)))  float    v8f;

// ------------------------------ small helpers ------------------------------

__device__ __forceinline__ float block_reduce_sum256(float v, float* red) {
  int tid = threadIdx.x;
  red[tid] = v; __syncthreads();
  #pragma unroll
  for (int s = 128; s > 0; s >>= 1) {
    if (tid < s) red[tid] += red[tid + s];
    __syncthreads();
  }
  float r = red[0]; __syncthreads();
  return r;
}

__device__ __forceinline__ float block_reduce_max256(float v, float* red) {
  int tid = threadIdx.x;
  red[tid] = v; __syncthreads();
  #pragma unroll
  for (int s = 128; s > 0; s >>= 1) {
    if (tid < s) red[tid] = fmaxf(red[tid], red[tid + s]);
    __syncthreads();
  }
  float r = red[0]; __syncthreads();
  return r;
}

// CDNA5 async global->LDS 16B copy (per-lane), tracked by ASYNCcnt.
__device__ __forceinline__ void async_copy_b128(unsigned lds_byte_off,
                                                const _Float16* gptr) {
  asm volatile("global_load_async_to_lds_b128 %0, %1, off"
               :: "v"(lds_byte_off),
                  "v"((unsigned long long)(__SIZE_TYPE__)gptr)
               : "memory");
}

// ------------------------- 1) squared norms of x ---------------------------
__global__ void xsq_kernel(const float* __restrict__ x, float* __restrict__ xsq) {
  int i = blockIdx.x * blockDim.x + threadIdx.x;       // over B*N
  if (i >= BB * NN) return;
  int b = i / NN, n = i % NN;
  const float* xp = x + (size_t)b * CC * NN + n;
  float s = 0.f;
  #pragma unroll
  for (int c = 0; c < CC; ++c) { float v = xp[(size_t)c * NN]; s += v * v; }
  xsq[i] = s;
}

// --------- 2) pairwise dist[b,n,m] = 2*x_n.x_m - |x_n|^2 - |x_m|^2 ---------
__global__ __launch_bounds__(256)
void dist_kernel(const float* __restrict__ x, const float* __restrict__ xsq,
                 float* __restrict__ dist) {
  __shared__ float xa[64][64];   // [c][n_local]
  __shared__ float xb[64][64];   // [c][m_local]
  int n0 = blockIdx.x * 64, m0 = blockIdx.y * 64, b = blockIdx.z;
  int tid = threadIdx.x;
  const float* xbase = x + (size_t)b * CC * NN;
  for (int i = tid; i < 64 * 64; i += 256) {
    int c = i >> 6, j = i & 63;
    xa[c][j] = (n0 + j < NN) ? xbase[(size_t)c * NN + n0 + j] : 0.f;
    xb[c][j] = (m0 + j < NN) ? xbase[(size_t)c * NN + m0 + j] : 0.f;
  }
  __syncthreads();
  int tx = tid & 15, ty = tid >> 4;
  float acc[4][4] = {};
  #pragma unroll
  for (int c = 0; c < 64; ++c) {
    float av[4], bv[4];
    #pragma unroll
    for (int i = 0; i < 4; ++i) av[i] = xa[c][ty * 4 + i];
    #pragma unroll
    for (int j = 0; j < 4; ++j) bv[j] = xb[c][tx * 4 + j];
    #pragma unroll
    for (int i = 0; i < 4; ++i)
      #pragma unroll
      for (int j = 0; j < 4; ++j) acc[i][j] += av[i] * bv[j];
  }
  const float* sq = xsq + (size_t)b * NN;
  #pragma unroll
  for (int i = 0; i < 4; ++i) {
    int n = n0 + ty * 4 + i;
    if (n >= NN) continue;
    #pragma unroll
    for (int j = 0; j < 4; ++j) {
      int m = m0 + tx * 4 + j;
      if (m >= NN) continue;
      dist[((size_t)b * NN + n) * NN + m] = 2.f * acc[i][j] - sq[n] - sq[m];
    }
  }
}

// ------------- 3) top-50 (descending, tie -> lower index) per row ----------
__global__ __launch_bounds__(128)
void topk_kernel(const float* __restrict__ dist, int* __restrict__ knn) {
  __shared__ float val[NN];
  __shared__ float rv[128];
  __shared__ int   ri[128];
  int row = blockIdx.x;                 // b*NN + n
  int tid = threadIdx.x;
  const float* dr = dist + (size_t)row * NN;
  for (int i = tid; i < NN; i += 128) val[i] = dr[i];
  __syncthreads();
  for (int kt = 0; kt < KT; ++kt) {
    float bv = -3.0e38f; int bi = NN;
    for (int i = tid; i < NN; i += 128) {
      float v = val[i];
      if (v > bv || (v == bv && i < bi)) { bv = v; bi = i; }
    }
    rv[tid] = bv; ri[tid] = bi; __syncthreads();
    #pragma unroll
    for (int s = 64; s > 0; s >>= 1) {
      if (tid < s) {
        if (rv[tid + s] > rv[tid] ||
            (rv[tid + s] == rv[tid] && ri[tid + s] < ri[tid])) {
          rv[tid] = rv[tid + s]; ri[tid] = ri[tid + s];
        }
      }
      __syncthreads();
    }
    if (tid == 0) { knn[(size_t)row * KT + kt] = ri[0]; val[ri[0]] = -3.0e38f; }
    __syncthreads();
  }
}

// ---------------- 4) relative neighbor features feat[b,n,kt,c] -------------
__global__ void feat_kernel(const float* __restrict__ x, const int* __restrict__ knn,
                            float* __restrict__ feat) {
  int row = blockIdx.x;                 // b*NN + n
  int b = row / NN, n = row % NN;
  int tid = threadIdx.x;
  const float* xb = x + (size_t)b * CC * NN;
  for (int e = tid; e < KT * CC; e += 256) {
    int kt = e >> 6, c = e & 63;
    int j = knn[(size_t)row * KT + kt];
    feat[(size_t)row * (KT * CC) + e] = xb[(size_t)c * NN + j] - xb[(size_t)c * NN + n];
  }
}

// ---- 5) transpose feat -> x_wn[b,m,n]: f32 residual + zero-padded f16 -----
__global__ void xwn_kernel(const float* __restrict__ feat,
                           float* __restrict__ xwn32, _Float16* __restrict__ xwn16) {
  __shared__ float tile[32][33];        // [m_local][n_local]
  int n0 = blockIdx.x * 32, m0 = blockIdx.y * 32, b = blockIdx.z;
  int tx = threadIdx.x, ty = threadIdx.y;
  #pragma unroll
  for (int i = 0; i < 4; ++i) {
    int nl = ty * 4 + i, n = n0 + nl;
    int m = m0 + tx, kt = m >> 6, c = m & 63;     // contiguous c -> coalesced read
    float v = 0.f;
    if (n < NN) v = feat[(((size_t)b * NN + n) * KT + kt) * CC + c];
    tile[tx][nl] = v;
  }
  __syncthreads();
  #pragma unroll
  for (int i = 0; i < 4; ++i) {
    int ml = ty * 4 + i, m = m0 + ml, n = n0 + tx;
    float v = tile[ml][tx];
    if (n < NN) xwn32[((size_t)b * MM + m) * NN + n] = v;
    xwn16[((size_t)b * MM + m) * NP + n] = (n < NN) ? (_Float16)v : (_Float16)0.f;
  }
}

// ---- 6) weights: w16[w][j][n] = (f16) W[j][n], zero padded to 1024x1024 ---
// (K-contiguous B layout for the GEMM: element(k=n, j) = w16[j*NP + n])
__global__ void wpad_kernel(const float* __restrict__ Wq, const float* __restrict__ Wk,
                            const float* __restrict__ Wv, const float* __restrict__ Wd,
                            _Float16* __restrict__ w16) {
  int j = blockIdx.x, z = blockIdx.y;
  const float* W = (z == 0) ? Wq : (z == 1) ? Wk : (z == 2) ? Wv : Wd;
  _Float16* dst = w16 + (size_t)z * NP * NP + (size_t)j * NP;
  for (int n = threadIdx.x; n < NP; n += 256)
    dst[n] = (j < NN && n < NN) ? (_Float16)W[(size_t)j * NN + n] : (_Float16)0.f;
}

// ---------------------------------------------------------------------------
// 7) Batched WMMA GEMM:  C[m,j] = sum_k A[m,k] * Bm[j*ldb + k] + bias[j]
//    (both operands K-contiguous). 128x128 tile, BK=32, 8 waves (4x2),
//    each wave a 32x64 patch = 2x4 v_wmma_f32_16x16x32_f16 per K-step.
//    Double-buffered LDS fed by global_load_async_to_lds_b128 (ASYNCcnt):
//    tile t+1 streams in while tile t is consumed. A/B fragments are two
//    ds_load_b128 per lane per the CDNA5 ISA 7.12.2 16-bit A layout.
//    transC16: write f16 output transposed (C16[col*ldc16 + row]).
// ---------------------------------------------------------------------------
#define TILE_B 10240u           // one 128x40-half tile, bytes
#define BUF_B  20480u           // A tile + B tile, bytes
#define TILE_H 5120             // one tile in halfs
#define BUF_H  10240            // buffer stride in halfs

__global__ __launch_bounds__(256)
void gemm_wmma(const _Float16* __restrict__ A, long long strideA, int lda,
               const _Float16* __restrict__ Bm, long long strideB, int ldb,
               const float* __restrict__ bias,
               float* __restrict__ C32, long long strideC32, int ldc32,
               _Float16* __restrict__ C16, long long strideC16, int ldc16,
               int transC16, int Ncols, int Kdim) {
  extern __shared__ _Float16 smem[];    // 2 * (A tile + B tile) = 40960 B
  int bm = blockIdx.x, bn = blockIdx.y, bz = blockIdx.z;
  A  += (size_t)bz * strideA;
  Bm += (size_t)bz * strideB;
  if (C32) C32 += (size_t)bz * strideC32;
  if (C16) C16 += (size_t)bz * strideC16;

  int tid = threadIdx.x, wid = tid >> 5, lane = tid & 31;
  int wm = wid & 3, wn = wid >> 2;      // 4 waves along M, 2 along N
  int hs = (lane >> 4) * 8;             // K-half select (ISA fragment layout)
  int l15 = lane & 15;

  // staging coords: each thread copies 2x16B of A and 2x16B of B per tile
  int r = tid >> 2, ck = (tid & 3) * 8;          // r: 0..63, ck: {0,8,16,24}
  const _Float16* a0 = A  + (size_t)(bm * 128 + r)      * lda + ck;
  const _Float16* a1 = A  + (size_t)(bm * 128 + r + 64) * lda + ck;
  const _Float16* b0 = Bm + (size_t)(bn * 128 + r)      * ldb + ck;
  const _Float16* b1 = Bm + (size_t)(bn * 128 + r + 64) * ldb + ck;
  unsigned la0 = (unsigned)(r * 80 + ck * 2);              // byte offsets
  unsigned la1 = (unsigned)((r + 64) * 80 + ck * 2);
  unsigned lb0 = TILE_B + la0;
  unsigned lb1 = TILE_B + la1;

  v8f acc[2][4] = {};

  // prologue: stage tile 0 into buffer 0
  async_copy_b128(la0, a0);
  async_copy_b128(la1, a1);
  async_copy_b128(lb0, b0);
  async_copy_b128(lb1, b1);

  int buf = 0;
  for (int k0 = 0; k0 < Kdim; k0 += 32) {
    if (k0 + 32 < Kdim) {               // stage next tile into other buffer
      unsigned nb = (buf ^ 1) * BUF_B;
      async_copy_b128(nb + la0, a0 + k0 + 32);
      async_copy_b128(nb + la1, a1 + k0 + 32);
      async_copy_b128(nb + lb0, b0 + k0 + 32);
      async_copy_b128(nb + lb1, b1 + k0 + 32);
      // in-order completion: <=4 outstanding means current tile has landed
      asm volatile("s_wait_asynccnt 0x4" ::: "memory");
    } else {
      asm volatile("s_wait_asynccnt 0x0" ::: "memory");
    }
    __syncthreads();                    // make current tile visible to all

    const _Float16* pa = smem + buf * BUF_H;
    const _Float16* pb = pa + TILE_H;
    v16h af[2], bf[4];
    #pragma unroll
    for (int mi = 0; mi < 2; ++mi) {
      const _Float16* p = pa + (wm * 32 + mi * 16 + l15) * 40 + hs;
      *(f16x8*)&af[mi]         = *(const f16x8*)p;          // K = hs..hs+7
      *(((f16x8*)&af[mi]) + 1) = *(const f16x8*)(p + 16);   // K = 16+hs..
    }
    #pragma unroll
    for (int ni = 0; ni < 4; ++ni) {
      const _Float16* p = pb + (wn * 64 + ni * 16 + l15) * 40 + hs;
      *(f16x8*)&bf[ni]         = *(const f16x8*)p;
      *(((f16x8*)&bf[ni]) + 1) = *(const f16x8*)(p + 16);
    }
    #pragma unroll
    for (int mi = 0; mi < 2; ++mi)
      #pragma unroll
      for (int ni = 0; ni < 4; ++ni)
        acc[mi][ni] = __builtin_amdgcn_wmma_f32_16x16x32_f16(
            false, af[mi], false, bf[ni], (short)0, acc[mi][ni], false, false);

    __syncthreads();                    // all reads done before buf is reused
    buf ^= 1;
  }

  // epilogue: C/D layout -> lane<16: N=lane,M=r ; lane>=16: N=lane-16,M=8+r
  int rowbase = bm * 128 + wm * 32 + ((lane >> 4) * 8);
  int colbase = bn * 128 + wn * 64;
  #pragma unroll
  for (int mi = 0; mi < 2; ++mi) {
    #pragma unroll
    for (int ni = 0; ni < 4; ++ni) {
      int col = colbase + ni * 16 + l15;
      float bv = (bias && col < Ncols) ? bias[col] : 0.f;
      #pragma unroll
      for (int rr = 0; rr < 8; ++rr) {
        int row = rowbase + mi * 16 + rr;
        float v = acc[mi][ni][rr] + bv;
        if (C32 && col < Ncols) C32[(size_t)row * ldc32 + col] = v;
        if (C16) {
          _Float16 hv = (col < Ncols) ? (_Float16)v : (_Float16)0.f;
          if (transC16) C16[(size_t)col * ldc16 + row] = hv;
          else          C16[(size_t)row * ldc16 + col] = hv;
        }
      }
    }
  }
}

// ------------------- 8) softmax over M with 1/sqrt(N) scale ----------------
__global__ __launch_bounds__(256)
void softmax_kernel(const float* __restrict__ logits, _Float16* __restrict__ attn) {
  __shared__ float red[256];
  const float scale = 0.03162277660168379f;   // 1/sqrt(1000)
  size_t row = blockIdx.x;
  const float* L = logits + row * MM;
  _Float16* O = attn + row * MM;
  int tid = threadIdx.x;
  float mx = -3.0e38f;
  for (int i = tid; i < MM; i += 256) mx = fmaxf(mx, L[i]);
  mx = block_reduce_max256(mx, red);
  float sum = 0.f;
  for (int i = tid; i < MM; i += 256) sum += __expf((L[i] - mx) * scale);
  sum = block_reduce_sum256(sum, red);
  float inv = 1.f / sum;
  for (int i = tid; i < MM; i += 256)
    O[i] = (_Float16)(__expf((L[i] - mx) * scale) * inv);
}

// -------- 9) h = LN(src + resid)*g + b ; f32 out + padded f16 out ----------
__global__ __launch_bounds__(256)
void ln_kernel(const float* __restrict__ src, const float* __restrict__ resid,
               const float* __restrict__ g, const float* __restrict__ bta,
               float* __restrict__ dst32, _Float16* __restrict__ dst16) {
  __shared__ float red[256];
  size_t row = blockIdx.x;
  const float* s = src + row * NN;
  const float* r = resid + row * NN;
  int tid = threadIdx.x;
  float sum = 0.f;
  for (int i = tid; i < NN; i += 256) sum += s[i] + r[i];
  float mu = block_reduce_sum256(sum, red) * (1.f / NN);
  float vs = 0.f;
  for (int i = tid; i < NN; i += 256) { float d = s[i] + r[i] - mu; vs += d * d; }
  float var = block_reduce_sum256(vs, red) * (1.f / NN);
  float rstd = rsqrtf(var + 1e-5f);
  for (int i = tid; i < NP; i += 256) {
    if (i < NN) {
      float h = (s[i] + r[i] - mu) * rstd * g[i] + bta[i];
      dst32[row * NN + i] = h;
      if (dst16) dst16[row * NP + i] = (_Float16)h;
    } else if (dst16) {
      dst16[row * NP + i] = (_Float16)0.f;
    }
  }
}

// ---- 10) score y.Wfc+bfc, stable-select 20 smallest, gather -> [B,C,K,N] --
__global__ __launch_bounds__(64)
void score_select_kernel(const float* __restrict__ h2, const float* __restrict__ feat,
                         const float* __restrict__ Wfc, const float* __restrict__ bfc,
                         float* __restrict__ out) {
  __shared__ float sred[64];
  __shared__ float sc[KT];
  __shared__ int   sel[KSEL];
  int row = blockIdx.x;                 // b*NN + n
  int b = row / NN, n = row % NN;
  int tid = threadIdx.x;                // channel c
  float wf = Wfc[tid];
  const float* fr = feat + (size_t)row * (KT * CC);
  for (int kt = 0; kt < KT; ++kt) {
    float v = wf * (h2[((size_t)b * MM + kt * CC + tid) * NN + n] + fr[kt * CC + tid]);
    sred[tid] = v; __syncthreads();
    #pragma unroll
    for (int s = 32; s > 0; s >>= 1) {
      if (tid < s) sred[tid] += sred[tid + s];
      __syncthreads();
    }
    if (tid == 0) sc[kt] = sred[0] + bfc[0];
    __syncthreads();
  }
  if (tid == 0) {                       // stable selection of 20 smallest
    bool used[KT];
    for (int i = 0; i < KT; ++i) used[i] = false;
    for (int k = 0; k < KSEL; ++k) {
      int best = -1;
      for (int i = 0; i < KT; ++i)
        if (!used[i] && (best < 0 || sc[i] < sc[best])) best = i;
      used[best] = true; sel[k] = best;
    }
  }
  __syncthreads();
  #pragma unroll
  for (int k = 0; k < KSEL; ++k)
    out[(((size_t)b * CC + tid) * KSEL + k) * NN + n] = fr[sel[k] * CC + tid];
}

// ---------------------------------------------------------------------------
extern "C" void kernel_launch(void* const* d_in, const int* in_sizes, int n_in,
                              void* d_out, int out_size, void* d_ws, size_t ws_size,
                              hipStream_t stream) {
  (void)in_sizes; (void)n_in; (void)out_size; (void)ws_size;
  const float* x    = (const float*)d_in[0];
  const float* Wq   = (const float*)d_in[1];
  const float* bq   = (const float*)d_in[2];
  const float* Wk   = (const float*)d_in[3];
  const float* bk   = (const float*)d_in[4];
  const float* Wv   = (const float*)d_in[5];
  const float* bv   = (const float*)d_in[6];
  const float* Wd   = (const float*)d_in[7];
  const float* bd   = (const float*)d_in[8];
  const float* Wfc  = (const float*)d_in[9];
  const float* bfc  = (const float*)d_in[10];
  const float* ln1g = (const float*)d_in[11];
  const float* ln1b = (const float*)d_in[12];
  const float* ln2g = (const float*)d_in[13];
  const float* ln2b = (const float*)d_in[14];

  // workspace carve (~1.2 GB total; d32 aliases out32, h2 aliases logits)
  char* wsb = (char*)d_ws;
  size_t off = 0;
  auto take = [&](size_t bytes) -> void* {
    void* p = wsb + off;
    off = (off + bytes + 255) & ~(size_t)255;
    return p;
  };
  float*    xsq    = (float*)   take((size_t)BB * NN * 4);
  float*    dist   = (float*)   take((size_t)BB * NN * NN * 4);
  int*      knn    = (int*)     take((size_t)BB * NN * KT * 4);
  float*    feat   = (float*)   take((size_t)BB * NN * KT * CC * 4);
  float*    xwn32  = (float*)   take((size_t)BB * MM * NN * 4);
  _Float16* xwn16  = (_Float16*)take((size_t)BB * MM * NP * 2);
  _Float16* w16    = (_Float16*)take((size_t)4 * NP * NP * 2);
  _Float16* q16    = (_Float16*)take((size_t)BB * MM * NP * 2);
  _Float16* k16    = (_Float16*)take((size_t)BB * MM * NP * 2);
  _Float16* v16t   = (_Float16*)take((size_t)BB * NP * MM * 2);  // [n][p] per batch
  float*    logits = (float*)   take((size_t)BB * MM * MM * 4);
  _Float16* attn16 = (_Float16*)take((size_t)BB * MM * MM * 2);
  float*    out32  = (float*)   take((size_t)BB * MM * NN * 4);
  float*    h132   = (float*)   take((size_t)BB * MM * NN * 4);
  _Float16* h116   = (_Float16*)take((size_t)BB * MM * NP * 2);
  float*    d32    = out32;     // out32 dead after ln1
  float*    h232   = logits;    // logits dead after softmax

  const long long sMN = (long long)MM * NP;   // f16 [M,NP] batch stride
  const long long sMM = (long long)MM * MM;   // [M,M] batch stride
  const long long sMn = (long long)MM * NN;   // f32 [M,N] batch stride
  const size_t    gemmLds = 2 * BUF_B;        // 40960 B double buffer

  // ---- KNN graph ----
  xsq_kernel<<<dim3((BB * NN + 255) / 256), 256, 0, stream>>>(x, xsq);
  dist_kernel<<<dim3(16, 16, BB), 256, 0, stream>>>(x, xsq, dist);
  topk_kernel<<<dim3(BB * NN), 128, 0, stream>>>(dist, knn);
  feat_kernel<<<dim3(BB * NN), 256, 0, stream>>>(x, knn, feat);
  xwn_kernel<<<dim3(32, 100, BB), dim3(32, 8), 0, stream>>>(feat, xwn32, xwn16);
  wpad_kernel<<<dim3(NP, 4), 256, 0, stream>>>(Wq, Wk, Wv, Wd, w16);

  // ---- q/k/v projections: B = W[j][n] (K-contiguous). v written transposed.
  gemm_wmma<<<dim3(MM / 128, NP / 128, BB), 256, gemmLds, stream>>>(
      xwn16, sMN, NP, w16 + 0 * (size_t)NP * NP, 0, NP, bq,
      nullptr, 0, 0, q16, sMN, NP, 0, NN, NP);
  gemm_wmma<<<dim3(MM / 128, NP / 128, BB), 256, gemmLds, stream>>>(
      xwn16, sMN, NP, w16 + 1 * (size_t)NP * NP, 0, NP, bk,
      nullptr, 0, 0, k16, sMN, NP, 0, NN, NP);
  gemm_wmma<<<dim3(MM / 128, NP / 128, BB), 256, gemmLds, stream>>>(
      xwn16, sMN, NP, w16 + 2 * (size_t)NP * NP, 0, NP, bv,
      nullptr, 0, 0, v16t, sMN, MM, 1 /*transC16*/, NN, NP);

  // ---- logits = q @ k^T : B rows are k16 token rows (K-contiguous) ----
  gemm_wmma<<<dim3(MM / 128, MM / 128, BB), 256, gemmLds, stream>>>(
      q16, sMN, NP, k16, sMN, NP, nullptr,
      logits, sMM, MM, nullptr, 0, 0, 0, MM, NP);

  softmax_kernel<<<dim3(BB * MM), 256, 0, stream>>>(logits, attn16);

  // ---- out = attn @ v : B rows are v16t[n][p] (K=p contiguous) ----
  gemm_wmma<<<dim3(MM / 128, NP / 128, BB), 256, gemmLds, stream>>>(
      attn16, sMM, MM, v16t, sMN, MM, nullptr,
      out32, sMn, NN, nullptr, 0, 0, 0, NN, MM);

  ln_kernel<<<dim3(BB * MM), 256, 0, stream>>>(out32, xwn32, ln1g, ln1b, h132, h116);

  // ---- d = h1 @ Wd^T + bd ----
  gemm_wmma<<<dim3(MM / 128, NP / 128, BB), 256, gemmLds, stream>>>(
      h116, sMN, NP, w16 + 3 * (size_t)NP * NP, 0, NP, bd,
      d32, sMn, NN, nullptr, 0, 0, 0, NN, NP);

  ln_kernel<<<dim3(BB * MM), 256, 0, stream>>>(d32, h132, ln2g, ln2b, h232, nullptr);

  score_select_kernel<<<dim3(BB * NN), 64, 0, stream>>>(
      h232, feat, Wfc, bfc, (float*)d_out);
}
